// AtomConvLayer_9929964388798
// MI455X (gfx1250) — matrix operand order repaired and checked by reading, not compiled, not run on hardware
//
#include <hip/hip_runtime.h>
#include <hip/hip_bf16.h>

typedef __attribute__((ext_vector_type(16))) _Float16 v16h;
typedef __attribute__((ext_vector_type(8)))  float    v8f;

#define NN   50000
#define MM   24
#define AA   64
#define BB   32
#define FAN  160   // 2A + B
#define HC   128   // 2A
#define EPSV 1e-5f

// workspace layout (bytes), all offsets 256-aligned
#define WH_OFF     0u           // 128*160 f16 = 40960 B
#define STATS_OFF  40960u       // 768 f32 stats (+pad) = 4096 B
#define GATED_OFF  45056u       // 50000*160 f16 = 16,000,000 B
#define H_OFF      16045056u    // 50000*128 f32 = 25,600,000 B
// stats region (float index): [0,128) sum1 [128,256) sumsq1 [256,384) mean1
// [384,512) rstd1 [512,576) sum2 [576,640) sumsq2 [640,704) mean2 [704,768) rstd2

__device__ __forceinline__ float softplusf(float x) {
    return fmaxf(x, 0.f) + log1pf(__expf(-fabsf(x)));
}

// ---------------- K0: convert W to f16, zero the stats scratch ----------------
__global__ void k0_prep(const float* __restrict__ W, _Float16* __restrict__ Wh,
                        float* __restrict__ stats) {
    int i = blockIdx.x * blockDim.x + threadIdx.x;
    if (i < HC * FAN) Wh[i] = (_Float16)W[i];
    if (i < 768) stats[i] = 0.f;
}

// ---------------- K1: fused gate weights + gather + neighbor/bond reduce ------
// one wave32 per atom; 8 atoms per 256-thread block; grid = 6250 (exact)
__global__ __launch_bounds__(256)
void k1_gate(const float* __restrict__ atom, const float* __restrict__ nbrf,
             const long long* __restrict__ idx, const float* __restrict__ bwi,
             const float* __restrict__ bwj, _Float16* __restrict__ gated) {
    __shared__ float s_w[8][MM];
    const int wv = threadIdx.x >> 5, lane = threadIdx.x & 31;
    const int n = blockIdx.x * 8 + wv;   // always < NN (6250*8 == 50000)

    if (lane < MM) s_w[wv][lane] = bwi[n * MM + lane] * bwj[n * MM + lane];
    __syncthreads();

    float wsum = 0.f;
#pragma unroll
    for (int m = 0; m < MM; ++m) wsum += s_w[wv][m];

    _Float16* grow = gated + (size_t)n * FAN;
    const float* arow = atom + (size_t)n * AA;

    // self_part + nbr_part: lanes cover a = lane, lane+32
#pragma unroll
    for (int r = 0; r < 2; ++r) {
        const int a = lane + 32 * r;
        grow[a] = (_Float16)(arow[a] * wsum);
        float acc = 0.f;
#pragma unroll 4
        for (int m = 0; m < MM; ++m) {
            const long long j = idx[n * MM + m];            // gather (L2-resident)
            acc += s_w[wv][m] * atom[(size_t)j * AA + a];
        }
        grow[AA + a] = (_Float16)acc;
    }
    // bond_part: lanes cover b = lane (0..31)
    {
        float acc = 0.f;
        const float* nrow = nbrf + (size_t)n * MM * BB;
#pragma unroll 4
        for (int m = 0; m < MM; ++m) acc += s_w[wv][m] * nrow[m * BB + lane];
        grow[2 * AA + lane] = (_Float16)acc;
    }
}

// ---------------- K2: WMMA GEMM  h = gated @ W^T + b  (+ bn1 partial stats) ---
// block = 256 threads = 8 waves; block handles 16 rows x 128 cols; grid = 3125
__global__ __launch_bounds__(256)
void k2_gemm(const _Float16* __restrict__ gated, const _Float16* __restrict__ Wh,
             const float* __restrict__ bias, float* __restrict__ h,
             float* __restrict__ stats) {
    __shared__ _Float16 s_g[16 * FAN];   // 5 KB tile of gated rows
    const int tid = threadIdx.x;
    const int rowbase = blockIdx.x * 16;

    for (int i = tid; i < 16 * FAN; i += 256)
        s_g[i] = gated[(size_t)rowbase * FAN + i];
    __syncthreads();

    const int wv = tid >> 5, lane = tid & 31;
    const int nloc = lane & 15, half = lane >> 4;
    const int c = wv * 16 + nloc;        // output column of this lane
    const int m = nloc;                  // A-matrix row of this lane

    v8f acc = {};
#pragma unroll
    for (int kb = 0; kb < FAN; kb += 32) {
        // A fragment: contiguous K per lane half (k = 16*half + element)
        v16h Av;
#pragma unroll
        for (int j = 0; j < 8; ++j) {
            const int kA = 2 * j + 16 * half + kb;
            Av[2 * j]     = s_g[m * FAN + kA];
            Av[2 * j + 1] = s_g[m * FAN + kA + 1];
        }
        // B fragment: W^T column c, contiguous 16 halfs (32B aligned: 320|kb*2|32)
        const v16h Bv = *(const v16h*)(Wh + (size_t)c * FAN + kb + 16 * half);
        acc = __builtin_amdgcn_wmma_f32_16x16x32_f16(
            /*neg_a=*/false, Av, /*neg_b=*/false, Bv,
            /*c_mod=*/(short)0, acc, /*reuse_a=*/false, /*reuse_b=*/false);
    }

    const float bc = bias[c];
    float lsum = 0.f, lsq = 0.f;
#pragma unroll
    for (int j = 0; j < 8; ++j) {        // D: VGPR j -> row j + 8*half
        const int row = rowbase + j + 8 * half;
        const float v = acc[j] + bc;
        h[(size_t)row * HC + c] = v;
        lsum += v; lsq += v * v;
    }
    lsum += __shfl_xor(lsum, 16, 32);    // combine the two lane-halves (same c)
    lsq  += __shfl_xor(lsq, 16, 32);
    if (half == 0) {
        atomicAdd(&stats[c], lsum);
        atomicAdd(&stats[HC + c], lsq);
    }
}

// ---------------- K3: finalize bn1 stats ----------------
__global__ void k3_fin1(float* __restrict__ stats) {
    const int t = threadIdx.x;           // 128
    const float mean = stats[t] / (float)NN;
    const float var  = stats[HC + t] / (float)NN - mean * mean;
    stats[256 + t] = mean;
    stats[384 + t] = rsqrtf(var + EPSV);
}

// ---------------- K4: bn1 + sigmoid*softplus gating (+ bn2 partial stats) -----
// 64 threads = column id; grid-stride over rows; 1 atomic pair per thread/block
__global__ __launch_bounds__(64)
void k4_act(const float* __restrict__ h, const float* __restrict__ g1,
            const float* __restrict__ be1, float* __restrict__ out,
            float* __restrict__ stats) {
    const int t = threadIdx.x;           // 0..63
    const float meanF = stats[256 + t],      rstdF = stats[384 + t];
    const float meanG = stats[256 + 64 + t], rstdG = stats[384 + 64 + t];
    const float gaF = g1[t], beF = be1[t], gaG = g1[64 + t], beG = be1[64 + t];
    float lsum = 0.f, lsq = 0.f;
    for (int n = blockIdx.x; n < NN; n += gridDim.x) {
        const float f = gaF * (h[(size_t)n * HC + t]      - meanF) * rstdF + beF;
        const float g = gaG * (h[(size_t)n * HC + 64 + t] - meanG) * rstdG + beG;
        const float core = (1.f / (1.f + __expf(-f))) * softplusf(g);
        out[(size_t)n * AA + t] = core;
        lsum += core; lsq += core * core;
    }
    atomicAdd(&stats[512 + t], lsum);
    atomicAdd(&stats[576 + t], lsq);
}

// ---------------- K5: finalize bn2 stats ----------------
__global__ void k5_fin2(float* __restrict__ stats) {
    const int t = threadIdx.x;           // 64
    const float mean = stats[512 + t] / (float)NN;
    const float var  = stats[576 + t] / (float)NN - mean * mean;
    stats[640 + t] = mean;
    stats[704 + t] = rsqrtf(var + EPSV);
}

// ---------------- K6: in-place bn2 + softplus ----------------
__global__ void k6_out(float* __restrict__ out, const float* __restrict__ g2,
                       const float* __restrict__ be2, const float* __restrict__ stats) {
    const int i = blockIdx.x * blockDim.x + threadIdx.x;
    if (i >= NN * AA) return;
    const int c = i & 63;
    const float y = g2[c] * (out[i] - stats[640 + c]) * stats[704 + c] + be2[c];
    out[i] = softplusf(y);
}

extern "C" void kernel_launch(void* const* d_in, const int* in_sizes, int n_in,
                              void* d_out, int out_size, void* d_ws, size_t ws_size,
                              hipStream_t stream) {
    const float*     atom = (const float*)d_in[0];
    const float*     nbrf = (const float*)d_in[1];
    const long long* idx  = (const long long*)d_in[2];   // int64
    const float*     bwi  = (const float*)d_in[3];
    const float*     bwj  = (const float*)d_in[4];
    const float*     W    = (const float*)d_in[5];
    const float*     bias = (const float*)d_in[6];
    const float*     g1   = (const float*)d_in[7];
    const float*     be1  = (const float*)d_in[8];
    const float*     g2   = (const float*)d_in[9];
    const float*     be2  = (const float*)d_in[10];

    char* ws = (char*)d_ws;
    _Float16* Wh    = (_Float16*)(ws + WH_OFF);
    float*    stats = (float*)(ws + STATS_OFF);
    _Float16* gated = (_Float16*)(ws + GATED_OFF);
    float*    h     = (float*)(ws + H_OFF);
    float*    out   = (float*)d_out;

    hipLaunchKernelGGL(k0_prep, dim3(80),    dim3(256), 0, stream, W, Wh, stats);
    hipLaunchKernelGGL(k1_gate, dim3(6250),  dim3(256), 0, stream, atom, nbrf, idx, bwi, bwj, gated);
    hipLaunchKernelGGL(k2_gemm, dim3(3125),  dim3(256), 0, stream, gated, Wh, bias, h, stats);
    hipLaunchKernelGGL(k3_fin1, dim3(1),     dim3(128), 0, stream, stats);
    hipLaunchKernelGGL(k4_act,  dim3(512),   dim3(64),  0, stream, h, g1, be1, out, stats);
    hipLaunchKernelGGL(k5_fin2, dim3(1),     dim3(64),  0, stream, stats);
    hipLaunchKernelGGL(k6_out,  dim3((NN * AA + 255) / 256), dim3(256), 0, stream, out, g2, be2, stats);
}
